// DecoderLayer_42159398977628
// MI455X (gfx1250) — compile-verified, matrix operand order checked by reading, not run on hardware
//
#include <hip/hip_runtime.h>
#include <hip/hip_bf16.h>
#include <math.h>
#include <cstdint>

// ---------------- problem constants ----------------
constexpr int B_  = 2;
constexpr int S_  = 2048;
constexpr int H_  = 2048;
constexpr int NH_ = 16;
constexpr int HD_ = 128;
constexpr int FF_ = 8192;
constexpr float EPS_ = 1e-6f;

typedef __attribute__((ext_vector_type(16))) __bf16 v16bf;
typedef __attribute__((ext_vector_type(4)))  __bf16 v4bf;
typedef __attribute__((ext_vector_type(8)))  float  v8f;
typedef __attribute__((ext_vector_type(4)))  float  v4f;
typedef __attribute__((ext_vector_type(4)))  unsigned int v4u;
typedef int v4i __attribute__((vector_size(16)));   // builtin's expected pointee

// ---------------------------------------------------------------------------
// Async global->LDS 16-byte copy (CDNA5 GLOBAL_LOAD_ASYNC_TO_LDS_B128,
// ASYNCcnt-tracked), with synchronous b128 fallback if the builtin is absent.
// Probe-confirmed signature: (v4i* src, v4i* dst, imm offset, imm cpol).
// ---------------------------------------------------------------------------
#if defined(__HIP_DEVICE_COMPILE__) && __has_builtin(__builtin_amdgcn_global_load_async_to_lds_b128)
#define USE_ASYNC_LDS 1
#else
#define USE_ASYNC_LDS 0
#endif

static __device__ inline void copy16_g2l(const __bf16* gsrc, __bf16* ldst) {
#if USE_ASYNC_LDS
    __builtin_amdgcn_global_load_async_to_lds_b128(
        (v4i*)(uintptr_t)gsrc, (v4i*)(uintptr_t)ldst, 0, 0);
#else
    *(v4u*)ldst = *(const v4u*)gsrc;
#endif
}

static __device__ inline void wait_lds_copies() {
#if USE_ASYNC_LDS
#if __has_builtin(__builtin_amdgcn_s_wait_asynccnt)
    __builtin_amdgcn_s_wait_asynccnt(0);
#else
    asm volatile("s_wait_asynccnt 0x0" ::: "memory");
#endif
#endif
}

// ---------------------------------------------------------------------------
// WMMA fragment loader (CDNA5 16x16x32 bf16, wave32).
// smem tile layout [idx16][K] (idx16 = M for A, N for B): lane[3:0] selects
// idx16; lanes 0-15 take K {0..7,16..23}, lanes 16-31 take K {8..15,24..31}.
// Both 8-element runs contiguous + 16B aligned -> 2x ds_load_b128.
// ---------------------------------------------------------------------------
static __device__ inline v16bf load_frag(const __bf16* smem, int ld,
                                         int base16, int k_base, int lane) {
    int row  = base16 + (lane & 15);
    int koff = (lane < 16) ? 0 : 8;
    v16bf f;
#pragma unroll
    for (int e = 0; e < 8; ++e) f[e]     = smem[row * ld + k_base + koff + e];
#pragma unroll
    for (int e = 0; e < 8; ++e) f[8 + e] = smem[row * ld + k_base + 16 + koff + e];
    return f;
}

// ---------------------------------------------------------------------------
// Weight convert+transpose: w[K][N] fp32 -> wT[N][K] bf16 (tiled, coalesced)
// ---------------------------------------------------------------------------
__global__ __launch_bounds__(256) void transpose_w_kernel(const float* __restrict__ w,
                                                          __bf16* __restrict__ wT,
                                                          int K, int N) {
    __shared__ float ts[32][33];
    int n0 = blockIdx.x * 32;
    int k0 = blockIdx.y * 32;
    int tx = threadIdx.x & 31, ty = threadIdx.x >> 5;
#pragma unroll
    for (int j = 0; j < 4; ++j)
        ts[ty + j * 8][tx] = w[(size_t)(k0 + ty + j * 8) * N + (n0 + tx)];
    __syncthreads();
#pragma unroll
    for (int j = 0; j < 4; ++j)
        wT[(size_t)(n0 + ty + j * 8) * K + (k0 + tx)] = (__bf16)ts[tx][ty + j * 8];
}

// ---------------------------------------------------------------------------
// V head-transpose: v[B,S,NH*HD] bf16 -> vT[B,NH,HD,S] bf16
// ---------------------------------------------------------------------------
__global__ __launch_bounds__(256) void transpose_v_kernel(const __bf16* __restrict__ v,
                                                          __bf16* __restrict__ vT) {
    __shared__ __bf16 ts[32][33];
    int s0 = blockIdx.x * 32;
    int h0 = blockIdx.y * 32;          // hd within head
    int bh = blockIdx.z;               // b*NH + head
    int b = bh >> 4, head = bh & 15;
    int tx = threadIdx.x & 31, ty = threadIdx.x >> 5;
#pragma unroll
    for (int j = 0; j < 4; ++j)
        ts[ty + j * 8][tx] = v[(size_t)(b * S_ + s0 + ty + j * 8) * H_ + head * HD_ + h0 + tx];
    __syncthreads();
#pragma unroll
    for (int j = 0; j < 4; ++j)
        vT[((size_t)bh * HD_ + h0 + ty + j * 8) * S_ + (s0 + tx)] = ts[tx][ty + j * 8];
}

// ---------------------------------------------------------------------------
// RMSNorm: fp32 in, bf16 out; one block per row of H_ = 2048
// ---------------------------------------------------------------------------
__global__ __launch_bounds__(256) void rmsnorm_kernel(const float* __restrict__ x,
                                                      const float* __restrict__ g,
                                                      __bf16* __restrict__ out) {
    int row = blockIdx.x;
    int tid = threadIdx.x;
    const float* xr = x + (size_t)row * H_;
    v4f vals[2];
    float local = 0.f;
#pragma unroll
    for (int i = 0; i < 2; ++i) {
        vals[i] = *(const v4f*)(xr + (tid + i * 256) * 4);
#pragma unroll
        for (int j = 0; j < 4; ++j) local += vals[i][j] * vals[i][j];
    }
    __shared__ float red[256];
    red[tid] = local;
    __syncthreads();
    for (int s = 128; s > 0; s >>= 1) {
        if (tid < s) red[tid] += red[tid + s];
        __syncthreads();
    }
    float inv = rsqrtf(red[0] * (1.0f / H_) + EPS_);
    __bf16* orow = out + (size_t)row * H_;
#pragma unroll
    for (int i = 0; i < 2; ++i) {
        int c = (tid + i * 256) * 4;
        v4f gv = *(const v4f*)(g + c);
        v4bf o;
#pragma unroll
        for (int j = 0; j < 4; ++j) o[j] = (__bf16)(gv[j] * vals[i][j] * inv);
        *(v4bf*)(orow + c) = o;
    }
}

// ---------------------------------------------------------------------------
// GEMM: C[M,N] = A[M,K] @ WT[N,K]^T, bf16 operands, f32 WMMA accumulate.
// Block tile 128x64, BK=32; 8 waves, wave w -> M-tile w, N-tiles 0..3.
// Both LDS tiles are contiguous 16B block copies (async global->LDS).
// ---------------------------------------------------------------------------
template <bool OUT_BF16>
__global__ __launch_bounds__(256) void gemm_bf16_kernel(const __bf16* __restrict__ A,
                                                        const __bf16* __restrict__ WT,
                                                        void* __restrict__ Cv,
                                                        int M, int N, int K) {
    __shared__ __bf16 As[128 * 32];  // [M][K]
    __shared__ __bf16 Bs[64 * 32];   // [N][K]
    int tid  = threadIdx.x;
    int lane = tid & 31;
    int wave = tid >> 5;
    int m0 = blockIdx.y * 128;
    int n0 = blockIdx.x * 64;

    v8f zero = {};
    v8f acc[4];
#pragma unroll
    for (int nt = 0; nt < 4; ++nt) acc[nt] = zero;

    for (int k0 = 0; k0 < K; k0 += 32) {
        __syncthreads();
#pragma unroll
        for (int i = 0; i < 2; ++i) {   // A: 512 x 16B
            int idx = i * 256 + tid;
            int r = idx >> 2, c8 = (idx & 3) * 8;
            copy16_g2l(A + (size_t)(m0 + r) * K + (k0 + c8), As + r * 32 + c8);
        }
        {                                // B: 256 x 16B
            int r = tid >> 2, c8 = (tid & 3) * 8;
            copy16_g2l(WT + (size_t)(n0 + r) * K + (k0 + c8), Bs + r * 32 + c8);
        }
        wait_lds_copies();
        __syncthreads();

        v16bf af = load_frag(As, 32, wave * 16, 0, lane);
#pragma unroll
        for (int nt = 0; nt < 4; ++nt) {
            v16bf bfb = load_frag(Bs, 32, nt * 16, 0, lane);
            acc[nt] = __builtin_amdgcn_wmma_f32_16x16x32_bf16(
                false, af, false, bfb, (short)0, acc[nt], false, false);
        }
    }

    int colb   = lane & 15;
    int rowadd = (lane < 16) ? 0 : 8;
#pragma unroll
    for (int nt = 0; nt < 4; ++nt)
#pragma unroll
        for (int r = 0; r < 8; ++r) {
            int row = m0 + wave * 16 + r + rowadd;
            int col = n0 + nt * 16 + colb;
            if (OUT_BF16)
                ((__bf16*)Cv)[(size_t)row * N + col] = (__bf16)acc[nt][r];
            else
                ((float*)Cv)[(size_t)row * N + col] = acc[nt][r];
        }
}

// ---------------------------------------------------------------------------
// Fused gate/up GEMM with SwiGLU epilogue: C = silu(A@Wg) * (A@Wu), bf16 out
// ---------------------------------------------------------------------------
__global__ __launch_bounds__(256) void gemm_gateup_kernel(const __bf16* __restrict__ A,
                                                          const __bf16* __restrict__ WgT,
                                                          const __bf16* __restrict__ WuT,
                                                          __bf16* __restrict__ C,
                                                          int M, int N, int K) {
    __shared__ __bf16 As[128 * 32];
    __shared__ __bf16 Gs[64 * 32];
    __shared__ __bf16 Us[64 * 32];
    int tid  = threadIdx.x;
    int lane = tid & 31;
    int wave = tid >> 5;
    int m0 = blockIdx.y * 128;
    int n0 = blockIdx.x * 64;

    v8f zero = {};
    v8f accg[4], accu[4];
#pragma unroll
    for (int nt = 0; nt < 4; ++nt) { accg[nt] = zero; accu[nt] = zero; }

    for (int k0 = 0; k0 < K; k0 += 32) {
        __syncthreads();
#pragma unroll
        for (int i = 0; i < 2; ++i) {
            int idx = i * 256 + tid;
            int r = idx >> 2, c8 = (idx & 3) * 8;
            copy16_g2l(A + (size_t)(m0 + r) * K + (k0 + c8), As + r * 32 + c8);
        }
        {
            int r = tid >> 2, c8 = (tid & 3) * 8;
            copy16_g2l(WgT + (size_t)(n0 + r) * K + (k0 + c8), Gs + r * 32 + c8);
            copy16_g2l(WuT + (size_t)(n0 + r) * K + (k0 + c8), Us + r * 32 + c8);
        }
        wait_lds_copies();
        __syncthreads();

        v16bf af = load_frag(As, 32, wave * 16, 0, lane);
#pragma unroll
        for (int nt = 0; nt < 4; ++nt) {
            v16bf gb = load_frag(Gs, 32, nt * 16, 0, lane);
            accg[nt] = __builtin_amdgcn_wmma_f32_16x16x32_bf16(
                false, af, false, gb, (short)0, accg[nt], false, false);
            v16bf ub = load_frag(Us, 32, nt * 16, 0, lane);
            accu[nt] = __builtin_amdgcn_wmma_f32_16x16x32_bf16(
                false, af, false, ub, (short)0, accu[nt], false, false);
        }
    }

    int colb   = lane & 15;
    int rowadd = (lane < 16) ? 0 : 8;
#pragma unroll
    for (int nt = 0; nt < 4; ++nt)
#pragma unroll
        for (int r = 0; r < 8; ++r) {
            int row = m0 + wave * 16 + r + rowadd;
            int col = n0 + nt * 16 + colb;
            float g = accg[nt][r];
            float u = accu[nt][r];
            float silu = g / (1.0f + __expf(-g));
            C[(size_t)row * N + col] = (__bf16)(silu * u);
        }
}

// ---------------------------------------------------------------------------
// Flash attention, causal. q,k: [B,S,NH*HD] bf16; vT: [B,NH,HD,S] bf16;
// out: [B,S,NH*HD] bf16. Block = (128 q-rows, head, batch); 8 waves.
// All three LDS operand tiles are contiguous async 16B copies.
// ---------------------------------------------------------------------------
__global__ __launch_bounds__(256) void flash_attn_kernel(const __bf16* __restrict__ q,
                                                         const __bf16* __restrict__ k,
                                                         const __bf16* __restrict__ vT,
                                                         __bf16* __restrict__ o) {
    __shared__ __bf16 Qs[128 * 128];   // 32 KB  [qrow][hd]
    __shared__ __bf16 Ks[32 * 128];    //  8 KB  [key][hd]
    __shared__ __bf16 Vts[128 * 32];   //  8 KB  [hd][key]
    __shared__ __bf16 Ps[8 * 16 * 32]; //  8 KB  per-wave [qrow][key]

    int tid  = threadIdx.x;
    int lane = tid & 31;
    int wave = tid >> 5;
    int qb   = blockIdx.x;
    int head = blockIdx.y;
    int b    = blockIdx.z;
    int q0   = qb * 128;

    const float scale = 0.08838834764831845f;  // 1/sqrt(128)

    const __bf16* qbase = q + ((size_t)b * S_ + q0) * H_ + head * HD_;
#pragma unroll
    for (int i = 0; i < 8; ++i) {              // 2048 x 16B
        int idx = i * 256 + tid;
        int r = idx >> 4, c8 = (idx & 15) * 8;
        copy16_g2l(qbase + (size_t)r * H_ + c8, Qs + r * 128 + c8);
    }

    float m_row[8], l_row[8];
    v8f zero = {};
    v8f oacc[8];
#pragma unroll
    for (int r = 0; r < 8; ++r) { m_row[r] = -1e30f; l_row[r] = 0.f; }
#pragma unroll
    for (int od = 0; od < 8; ++od) oacc[od] = zero;

    int colb   = lane & 15;
    int rowadd = (lane < 16) ? 0 : 8;
    int nkb    = (q0 + 128) / 32;

    for (int kb = 0; kb < nkb; ++kb) {
        int kstart = kb * 32;
        __syncthreads();
        const __bf16* kbase  = k  + ((size_t)b * S_ + kstart) * H_ + head * HD_;
        const __bf16* vtbase = vT + ((size_t)(b * NH_ + head) * HD_) * S_ + kstart;
        {                                       // K: 256 x 16B
            int r = tid >> 4, c8 = (tid & 15) * 8;
            copy16_g2l(kbase + (size_t)r * H_ + c8, Ks + r * 128 + c8);
        }
#pragma unroll
        for (int i = 0; i < 2; ++i) {           // Vt: 512 x 16B
            int idx = i * 256 + tid;
            int hd = idx >> 2, c8 = (idx & 3) * 8;
            copy16_g2l(vtbase + (size_t)hd * S_ + c8, Vts + hd * 32 + c8);
        }
        wait_lds_copies();
        __syncthreads();

        // scores: S[16 x 32keys] per wave, 2 key-tiles of 16
        v8f s_acc[2];
        s_acc[0] = zero; s_acc[1] = zero;
#pragma unroll
        for (int c = 0; c < 4; ++c) {
            v16bf qf = load_frag(Qs, 128, wave * 16, c * 32, lane);
#pragma unroll
            for (int nt = 0; nt < 2; ++nt) {
                v16bf kf = load_frag(Ks, 128, nt * 16, c * 32, lane);
                s_acc[nt] = __builtin_amdgcn_wmma_f32_16x16x32_bf16(
                    false, qf, false, kf, (short)0, s_acc[nt], false, false);
            }
        }

        float sv[2][8];
#pragma unroll
        for (int nt = 0; nt < 2; ++nt)
#pragma unroll
            for (int r = 0; r < 8; ++r) {
                float x = s_acc[nt][r] * scale;
                int kcol = kstart + nt * 16 + colb;
                int qrow = q0 + wave * 16 + r + rowadd;
                sv[nt][r] = (kcol > qrow) ? -1e30f : x;
            }

        float mnew[8], alpha[8];
#pragma unroll
        for (int r = 0; r < 8; ++r) {
            float t = fmaxf(sv[0][r], sv[1][r]);
#pragma unroll
            for (int off = 8; off >= 1; off >>= 1)
                t = fmaxf(t, __shfl_xor(t, off, 16));
            mnew[r]  = fmaxf(m_row[r], t);
            alpha[r] = __expf(m_row[r] - mnew[r]);
            m_row[r] = mnew[r];
        }
#pragma unroll
        for (int nt = 0; nt < 2; ++nt)
#pragma unroll
            for (int r = 0; r < 8; ++r)
                sv[nt][r] = __expf(sv[nt][r] - mnew[r]);
#pragma unroll
        for (int r = 0; r < 8; ++r) {
            float t = sv[0][r] + sv[1][r];
#pragma unroll
            for (int off = 8; off >= 1; off >>= 1)
                t += __shfl_xor(t, off, 16);
            l_row[r] = alpha[r] * l_row[r] + t;
        }
#pragma unroll
        for (int od = 0; od < 8; ++od)
#pragma unroll
            for (int r = 0; r < 8; ++r)
                oacc[od][r] *= alpha[r];

        __bf16* pw = Ps + wave * 16 * 32;
#pragma unroll
        for (int nt = 0; nt < 2; ++nt)
#pragma unroll
            for (int r = 0; r < 8; ++r)
                pw[(r + rowadd) * 32 + nt * 16 + colb] = (__bf16)sv[nt][r];

        v16bf pf = load_frag(pw, 32, 0, 0, lane);
#pragma unroll
        for (int od = 0; od < 8; ++od) {
            v16bf vf = load_frag(Vts, 32, od * 16, 0, lane);
            oacc[od] = __builtin_amdgcn_wmma_f32_16x16x32_bf16(
                false, pf, false, vf, (short)0, oacc[od], false, false);
        }
    }

    __bf16* obase = o + ((size_t)b * S_ + q0) * H_ + head * HD_;
    float inv[8];
#pragma unroll
    for (int r = 0; r < 8; ++r) inv[r] = 1.0f / l_row[r];
#pragma unroll
    for (int od = 0; od < 8; ++od)
#pragma unroll
        for (int r = 0; r < 8; ++r) {
            int row = wave * 16 + r + rowadd;
            int col = od * 16 + colb;
            obase[(size_t)row * H_ + col] = (__bf16)(oacc[od][r] * inv[r]);
        }
}

// ---------------------------------------------------------------------------
// Elementwise residual add: c = a + b  (c may alias a)
// ---------------------------------------------------------------------------
__global__ __launch_bounds__(256) void add_kernel(const float* __restrict__ a,
                                                  const float* __restrict__ b,
                                                  float* __restrict__ c, size_t n4) {
    size_t idx = (size_t)blockIdx.x * 256 + threadIdx.x;
    if (idx < n4) {
        v4f av = *(const v4f*)(a + idx * 4);
        v4f bv = *(const v4f*)(b + idx * 4);
        v4f cv;
#pragma unroll
        for (int j = 0; j < 4; ++j) cv[j] = av[j] + bv[j];
        *(v4f*)(c + idx * 4) = cv;
    }
}

// ---------------------------------------------------------------------------
extern "C" void kernel_launch(void* const* d_in, const int* in_sizes, int n_in,
                              void* d_out, int out_size, void* d_ws, size_t ws_size,
                              hipStream_t stream) {
    (void)in_sizes; (void)n_in; (void)out_size; (void)ws_size;

    const float* x      = (const float*)d_in[0];
    // d_in[1] = dense causal mask (unused; causality applied analytically)
    const float* w_q    = (const float*)d_in[2];
    const float* w_k    = (const float*)d_in[3];
    const float* w_v    = (const float*)d_in[4];
    const float* w_o    = (const float*)d_in[5];
    const float* w_gate = (const float*)d_in[6];
    const float* w_up   = (const float*)d_in[7];
    const float* w_down = (const float*)d_in[8];
    const float* g_in   = (const float*)d_in[9];
    const float* g_post = (const float*)d_in[10];
    float* out = (float*)d_out;

    const int M = B_ * S_;                       // 4096
    const size_t HH2 = (size_t)H_ * H_ * 2;      // 8.4 MB  (bf16 HxH)
    const size_t HF2 = (size_t)H_ * FF_ * 2;     // 33.5 MB (bf16 HxFF)
    const size_t MH2 = (size_t)M * H_ * 2;       // 16.8 MB (bf16 MxH)

    char* p = (char*)d_ws;
    __bf16* wqT   = (__bf16*)p;            p += HH2;
    __bf16* wkT   = (__bf16*)p;            p += HH2;
    __bf16* wvT   = (__bf16*)p;            p += HH2;
    __bf16* woT   = (__bf16*)p;            p += HH2;
    __bf16* wgT   = (__bf16*)p;            p += HF2;
    __bf16* wuT   = (__bf16*)p;            p += HF2;
    __bf16* wdT   = (__bf16*)p;            p += HF2;
    __bf16* hb    = (__bf16*)p;            p += MH2;
    __bf16* qbuf  = (__bf16*)p;            p += MH2;
    __bf16* kbuf  = (__bf16*)p;            p += MH2;
    __bf16* vbuf  = (__bf16*)p;            p += MH2;
    __bf16* vT    = (__bf16*)p;            p += MH2;
    __bf16* attnb = (__bf16*)p;            p += MH2;
    float*  tmp   = (float*)p;             p += (size_t)M * H_ * 4;
    __bf16* actb  = qbuf;  // [M,FF] bf16 = 4*MH2, reuses qbuf..vT (dead after attention)

    dim3 blk(256);
    dim3 gemmH(H_ / 64, M / 128);    // (32, 32)
    dim3 gemmF(FF_ / 64, M / 128);   // (128, 32)
    dim3 attnG(S_ / 128, NH_, B_);   // (16, 16, 2)
    size_t n4 = (size_t)M * H_ / 4;
    int addBlocks = (int)((n4 + 255) / 256);

    // --- one-time (per call) weight convert+transpose: fp32 [K][N] -> bf16 [N][K]
    transpose_w_kernel<<<dim3(H_ / 32, H_ / 32), blk, 0, stream>>>(w_q, wqT, H_, H_);
    transpose_w_kernel<<<dim3(H_ / 32, H_ / 32), blk, 0, stream>>>(w_k, wkT, H_, H_);
    transpose_w_kernel<<<dim3(H_ / 32, H_ / 32), blk, 0, stream>>>(w_v, wvT, H_, H_);
    transpose_w_kernel<<<dim3(H_ / 32, H_ / 32), blk, 0, stream>>>(w_o, woT, H_, H_);
    transpose_w_kernel<<<dim3(FF_ / 32, H_ / 32), blk, 0, stream>>>(w_gate, wgT, H_, FF_);
    transpose_w_kernel<<<dim3(FF_ / 32, H_ / 32), blk, 0, stream>>>(w_up, wuT, H_, FF_);
    transpose_w_kernel<<<dim3(H_ / 32, FF_ / 32), blk, 0, stream>>>(w_down, wdT, FF_, H_);

    // --- attention block ---
    rmsnorm_kernel<<<M, blk, 0, stream>>>(x, g_in, hb);
    gemm_bf16_kernel<true><<<gemmH, blk, 0, stream>>>(hb, wqT, qbuf, M, H_, H_);
    gemm_bf16_kernel<true><<<gemmH, blk, 0, stream>>>(hb, wkT, kbuf, M, H_, H_);
    gemm_bf16_kernel<true><<<gemmH, blk, 0, stream>>>(hb, wvT, vbuf, M, H_, H_);
    transpose_v_kernel<<<dim3(S_ / 32, HD_ / 32, B_ * NH_), blk, 0, stream>>>(vbuf, vT);
    flash_attn_kernel<<<attnG, blk, 0, stream>>>(qbuf, kbuf, vT, attnb);
    gemm_bf16_kernel<false><<<gemmH, blk, 0, stream>>>(attnb, woT, tmp, M, H_, H_);
    add_kernel<<<addBlocks, blk, 0, stream>>>(x, tmp, out, n4);

    // --- MLP block (SwiGLU) ---
    rmsnorm_kernel<<<M, blk, 0, stream>>>(out, g_post, hb);
    gemm_gateup_kernel<<<gemmF, blk, 0, stream>>>(hb, wgT, wuT, actb, M, FF_, H_);
    gemm_bf16_kernel<false><<<gemmH, blk, 0, stream>>>(actb, wdT, tmp, M, H_, FF_);
    add_kernel<<<addBlocks, blk, 0, stream>>>(out, tmp, out, n4);
}
